// GraphEncoder_36867999269121
// MI455X (gfx1250) — matrix operand order repaired and checked by reading, not compiled, not run on hardware
//
#include <hip/hip_runtime.h>
#include <hip/hip_bf16.h>
#include <hip/hip_fp16.h>

typedef __attribute__((ext_vector_type(16))) _Float16 v16h;
typedef __attribute__((ext_vector_type(8)))  _Float16 v8h;
typedef __attribute__((ext_vector_type(8)))  float    v8f;

#define HID 64
#define HEADS 2
#define FDIM (HID * HEADS)   // 128
#define NEG_SLOPE 0.2f

// ---------- ordered-float encoding for atomic max on floats ----------
__device__ __forceinline__ unsigned f2ord(float f) {
  unsigned u = __float_as_uint(f);
  return (u & 0x80000000u) ? ~u : (u | 0x80000000u);
}
__device__ __forceinline__ float ord2f(unsigned u) {
  return (u & 0x80000000u) ? __uint_as_float(u & 0x7FFFFFFFu)
                           : __uint_as_float(~u);
}
#define ORD_NEG_INF 0x007FFFFFu   // f2ord(-inf)

__device__ __forceinline__ void edge_endpoints(int e, const int* __restrict__ ei,
                                               int nE, int n, int& src, int& dst) {
  if (e < nE) { src = ei[e]; dst = ei[nE + e]; }
  else        { src = e - nE; dst = e - nE; }   // self loops appended
}

// ---------- weight conversion fp32 -> f16 ----------
__global__ void k_cvt_weights(const float* __restrict__ W1, const float* __restrict__ W2,
                              const float* __restrict__ Wp,
                              _Float16* __restrict__ W1h, _Float16* __restrict__ W2h,
                              _Float16* __restrict__ Wph) {
  int i = blockIdx.x * blockDim.x + threadIdx.x;
  if (i < FDIM * HID)  W1h[i] = (_Float16)W1[i];
  if (i < FDIM * FDIM) W2h[i] = (_Float16)W2[i];
  if (i < HID * FDIM)  Wph[i] = (_Float16)Wp[i];
}

// ---------- embedding gather (int token -> f16 features) ----------
__global__ void k_embed(const int* __restrict__ x, const float* __restrict__ emb,
                        _Float16* __restrict__ h, int n) {
  int i = blockIdx.x * blockDim.x + threadIdx.x;
  if (i >= n * HID) return;
  int node = i / HID, k = i - node * HID;
  h[i] = (_Float16)emb[x[node] * HID + k];
}

// ---------- WMMA GEMM: D[n,OUT] = A[n,K](f16) * W[OUT,K](f16)^T (+bias) ----------
// One wave per 16x16 output tile, 4 waves per block covering 64 output cols.
// Optionally also writes a f16 mirror D16 (used to halve edge-gather traffic).
__global__ void k_wmma_gemm(const _Float16* __restrict__ A,
                            const _Float16* __restrict__ W,
                            const float* __restrict__ bias,
                            float* __restrict__ D,
                            _Float16* __restrict__ D16,
                            int n, int K, int OUT) {
  int lane = threadIdx.x & 31;
  int wave = threadIdx.x >> 5;
  int rowBase = blockIdx.x * 16;
  int colBase = (blockIdx.y * 4 + wave) * 16;
  if (colBase >= OUT) return;                 // wave-uniform
  int m = lane & 15, g = lane >> 4;

  int rowA = rowBase + m; if (rowA >= n) rowA = n - 1;   // keep EXEC all-1s
  const _Float16* ap = A + (size_t)rowA * K;
  const _Float16* bp = W + (size_t)(colBase + m) * K + 16 * g;  // B: lane = out col

  v8f c = {};
  for (int kb = 0; kb < K; kb += 32) {
    // A 16x32 f16 layout: elems e -> K = kb + 16g + e (one contiguous 32B load)
    v16h a = *(const v16h*)(ap + kb + 16 * g);
    // B 32x16 f16 layout: elems e -> K = kb + 16g + e (one contiguous 32B load)
    v16h b = *(const v16h*)(bp + kb);
    c = __builtin_amdgcn_wmma_f32_16x16x32_f16(false, a, false, b,
                                               (short)0, c, false, false);
  }

  int col = colBase + m;
  float bv = bias ? bias[col] : 0.0f;
#pragma unroll
  for (int v = 0; v < 8; ++v) {
    int row = rowBase + v + 8 * g;            // C/D layout: VGPR v -> row v + 8*(lane>>4)
    if (row < n) {
      float out = c[v] + bv;
      D[(size_t)row * OUT + col] = out;
      if (D16) D16[(size_t)row * OUT + col] = (_Float16)out;
    }
  }
}

// ---------- attention coefficients (wave per node, coalesced float4 row read) ----
// lanes 0..15 -> head 0 channels, lanes 16..31 -> head 1 channels (4 ch/lane);
// a_src/a_dst are flat [H*HID] so lane l reads a[4l..4l+3] directly.
__global__ void k_att(const float* __restrict__ xl, const float* __restrict__ a_src,
                      const float* __restrict__ a_dst,
                      float* __restrict__ alpha_s, float* __restrict__ alpha_d, int n) {
  int lane = threadIdx.x & 31;
  int node = blockIdx.x * (blockDim.x >> 5) + (threadIdx.x >> 5);
  if (node >= n) return;
  float4 xv  = *(const float4*)(xl + (size_t)node * FDIM + lane * 4);
  float4 as4 = *(const float4*)(a_src + lane * 4);
  float4 ad4 = *(const float4*)(a_dst + lane * 4);
  float ss = xv.x * as4.x + xv.y * as4.y + xv.z * as4.z + xv.w * as4.w;
  float sd = xv.x * ad4.x + xv.y * ad4.y + xv.z * ad4.z + xv.w * ad4.w;
#pragma unroll
  for (int msk = 1; msk < 16; msk <<= 1) {    // reduce within each 16-lane half
    ss += __shfl_xor(ss, msk);
    sd += __shfl_xor(sd, msk);
  }
  if ((lane & 15) == 0) {
    int h = lane >> 4;
    alpha_s[node * HEADS + h] = ss;
    alpha_d[node * HEADS + h] = sd;
  }
}

// ---------- per-layer state init ----------
__global__ void k_init(unsigned* __restrict__ m_enc, float* __restrict__ ssum,
                       float* __restrict__ acc, int n) {
  int i = blockIdx.x * blockDim.x + threadIdx.x;
  if (i < n * HEADS) { m_enc[i] = ORD_NEG_INF; ssum[i] = 0.f; }
  if (i < n * FDIM)  acc[i] = 0.f;
}

// ---------- edge pass 1: leaky-relu logits + segment max into dst ----------
__global__ void k_edge_max(const int* __restrict__ ei, int nE, int n,
                           const float* __restrict__ alpha_s, const float* __restrict__ alpha_d,
                           float* __restrict__ e_buf, unsigned* __restrict__ m_enc) {
  int idx = blockIdx.x * blockDim.x + threadIdx.x;
  int E2 = nE + n;
  if (idx >= E2 * HEADS) return;
  int e = idx >> 1, h = idx & 1;
  int src, dst; edge_endpoints(e, ei, nE, n, src, dst);
  float v = alpha_s[src * HEADS + h] + alpha_d[dst * HEADS + h];
  v = v > 0.f ? v : v * NEG_SLOPE;
  e_buf[idx] = v;
  atomicMax(m_enc + dst * HEADS + h, f2ord(v));
}

// ---------- edge pass 2: exp(e - m[dst]) + segment sum ----------
__global__ void k_edge_exp(const int* __restrict__ ei, int nE, int n,
                           const float* __restrict__ e_buf, const unsigned* __restrict__ m_enc,
                           float* __restrict__ ex_buf, float* __restrict__ ssum) {
  int idx = blockIdx.x * blockDim.x + threadIdx.x;
  int E2 = nE + n;
  if (idx >= E2 * HEADS) return;
  int e = idx >> 1, h = idx & 1;
  int src, dst; edge_endpoints(e, ei, nE, n, src, dst);
  (void)src;
  float ex = expf(e_buf[idx] - ord2f(m_enc[dst * HEADS + h]));
  ex_buf[idx] = ex;
  atomicAdd(ssum + dst * HEADS + h, ex);
}

// ---------- edge pass 3: weighted scatter acc[dst] += alpha * xl16[src] ----------
// 64 threads/edge, 2 channels per thread from the f16 mirror (256B/edge gather).
__global__ void k_edge_scatter(const int* __restrict__ ei, int nE, int n,
                               const _Float16* __restrict__ xl16,
                               const float* __restrict__ ex_buf,
                               const float* __restrict__ ssum, float* __restrict__ acc) {
  int e = blockIdx.x;
  int j = threadIdx.x * 2;                    // channel pair (never crosses head bound)
  int src, dst; edge_endpoints(e, ei, nE, n, src, dst);
  int h = j >> 6;
  float alpha = ex_buf[e * HEADS + h] / (ssum[dst * HEADS + h] + 1e-16f);
  const _Float16* xp = xl16 + (size_t)src * FDIM + j;
  float v0 = (float)xp[0], v1 = (float)xp[1];
  float* apc = acc + (size_t)dst * FDIM + j;
  atomicAdd(apc,     v0 * alpha);
  atomicAdd(apc + 1, v1 * alpha);
}

// ---------- node epilogue: + bias, ELU, convert to f16 for next GEMM ----------
__global__ void k_elu_cvt(const float* __restrict__ acc, const float* __restrict__ bias,
                          _Float16* __restrict__ out16, int n) {
  int i = blockIdx.x * blockDim.x + threadIdx.x;
  if (i >= n * FDIM) return;
  float v = acc[i] + bias[i & (FDIM - 1)];
  v = v > 0.f ? v : expm1f(v);               // ELU (alpha = 1)
  out16[i] = (_Float16)v;
}

extern "C" void kernel_launch(void* const* d_in, const int* in_sizes, int n_in,
                              void* d_out, int out_size, void* d_ws, size_t ws_size,
                              hipStream_t stream) {
  const int*   x   = (const int*)d_in[0];
  const int*   ei  = (const int*)d_in[1];     // [2, E] flat
  const float* emb = (const float*)d_in[2];
  const float* W1  = (const float*)d_in[3];
  const float* as1 = (const float*)d_in[4];
  const float* ad1 = (const float*)d_in[5];
  const float* b1  = (const float*)d_in[6];
  const float* W2  = (const float*)d_in[7];
  const float* as2 = (const float*)d_in[8];
  const float* ad2 = (const float*)d_in[9];
  const float* b2  = (const float*)d_in[10];
  const float* Wp  = (const float*)d_in[11];
  const float* bp  = (const float*)d_in[12];

  const int n  = in_sizes[0];
  const int nE = in_sizes[1] / 2;
  const int E2 = nE + n;

  // ---- carve workspace ----
  char* p = (char*)d_ws;
  auto carve = [&](size_t bytes) { void* r = (void*)p; p += (bytes + 255) & ~(size_t)255; return r; };
  _Float16* W1h   = (_Float16*)carve((size_t)FDIM * HID * 2);
  _Float16* W2h   = (_Float16*)carve((size_t)FDIM * FDIM * 2);
  _Float16* Wph   = (_Float16*)carve((size_t)HID * FDIM * 2);
  _Float16* h16   = (_Float16*)carve((size_t)n * FDIM * 2);
  _Float16* xl16  = (_Float16*)carve((size_t)n * FDIM * 2);
  float*    xl    = (float*)carve((size_t)n * FDIM * 4);
  float*    acc   = (float*)carve((size_t)n * FDIM * 4);
  float*    a_s   = (float*)carve((size_t)n * HEADS * 4);
  float*    a_d   = (float*)carve((size_t)n * HEADS * 4);
  unsigned* m_enc = (unsigned*)carve((size_t)n * HEADS * 4);
  float*    ssum  = (float*)carve((size_t)n * HEADS * 4);
  float*    e_buf = (float*)carve((size_t)E2 * HEADS * 4);
  float*    exbuf = (float*)carve((size_t)E2 * HEADS * 4);

  const int T = 256;
  const int nodeF  = (n * FDIM + T - 1) / T;
  const int nodeH  = (n * HID + T - 1) / T;
  const int nodeW  = (n + 7) / 8;            // 8 waves (nodes) per 256-thr block
  const int edgeT  = (E2 * HEADS + T - 1) / T;
  dim3 gemm128((n + 15) / 16, FDIM / 64);    // OUT=128 -> grid.y=2, 4 waves/block
  dim3 gemm64((n + 15) / 16, 1);             // OUT=64

  k_cvt_weights<<<(FDIM * FDIM + T - 1) / T, T, 0, stream>>>(W1, W2, Wp, W1h, W2h, Wph);
  k_embed<<<nodeH, T, 0, stream>>>(x, emb, h16, n);

  // ---- layer 1 ----
  k_wmma_gemm<<<gemm128, 128, 0, stream>>>(h16, W1h, nullptr, xl, xl16, n, HID, FDIM);
  k_att<<<nodeW, T, 0, stream>>>(xl, as1, ad1, a_s, a_d, n);
  k_init<<<nodeF, T, 0, stream>>>(m_enc, ssum, acc, n);
  k_edge_max<<<edgeT, T, 0, stream>>>(ei, nE, n, a_s, a_d, e_buf, m_enc);
  k_edge_exp<<<edgeT, T, 0, stream>>>(ei, nE, n, e_buf, m_enc, exbuf, ssum);
  k_edge_scatter<<<E2, FDIM / 2, 0, stream>>>(ei, nE, n, xl16, exbuf, ssum, acc);
  k_elu_cvt<<<nodeF, T, 0, stream>>>(acc, b1, h16, n);

  // ---- layer 2 ----
  k_wmma_gemm<<<gemm128, 128, 0, stream>>>(h16, W2h, nullptr, xl, xl16, n, FDIM, FDIM);
  k_att<<<nodeW, T, 0, stream>>>(xl, as2, ad2, a_s, a_d, n);
  k_init<<<nodeF, T, 0, stream>>>(m_enc, ssum, acc, n);
  k_edge_max<<<edgeT, T, 0, stream>>>(ei, nE, n, a_s, a_d, e_buf, m_enc);
  k_edge_exp<<<edgeT, T, 0, stream>>>(ei, nE, n, e_buf, m_enc, exbuf, ssum);
  k_edge_scatter<<<E2, FDIM / 2, 0, stream>>>(ei, nE, n, xl16, exbuf, ssum, acc);
  k_elu_cvt<<<nodeF, T, 0, stream>>>(acc, b2, h16, n);

  // ---- projection (fused bias) ----
  k_wmma_gemm<<<gemm64, 128, 0, stream>>>(h16, Wph, bp, (float*)d_out, nullptr, n, FDIM, HID);
}
